// MultiHeadAttentionQuantum_65481071396078
// MI455X (gfx1250) — matrix operand order repaired and checked by reading, not compiled, not run on hardware
//
#include <hip/hip_runtime.h>
#include <stdint.h>

typedef _Float16 f16;
typedef __attribute__((ext_vector_type(16))) _Float16 v16h;
typedef __attribute__((ext_vector_type(8)))  float    v8f;
typedef __attribute__((ext_vector_type(4)))  int      v4i;
typedef __attribute__((ext_vector_type(8)))  int      v8i;

#define B_DIM  2
#define S_DIM  2048
#define E_DIM  128
#define H_DIM  16
#define DK_DIM 8

// ---------- WMMA helper (probe-verified builtin) ----------
__device__ __forceinline__ v8f wmma_f16(v16h a, v16h b, v8f c) {
  // D = A(16x32 f16) * B(32x16 f16) + C(16x16 f32)
  return __builtin_amdgcn_wmma_f32_16x16x32_f16(false, a, false, b, (short)0, c,
                                                false, false);
}

// ---------- Operand gathers (layouts per CDNA5 ISA) ----------
// A (16x32, f16): lane 0-15 -> M=lane, K groups {0..7,16..23}; lanes 16-31 -> +8
__device__ __forceinline__ v16h gatherA_f16(const f16* base, int rstride, int kstride,
                                            int lane, int rvalid, int kvalid) {
  const int row = lane & 15;
  const int kg  = (lane >> 4) & 1;
  v16h a;
#pragma unroll
  for (int h = 0; h < 16; ++h) {
    const int k = ((h >> 3) << 4) + (kg << 3) + (h & 7);
    f16 v = (f16)0.f;
    if (row < rvalid && k < kvalid) v = base[row * rstride + k * kstride];
    a[h] = v;
  }
  return a;
}

// B (32x16, f16): lane 0-15 -> N=lane, K groups {0..7,16..23}; lanes 16-31 -> +8
__device__ __forceinline__ v16h gatherB_f16(const f16* base, int cstride, int kstride,
                                            int lane, int cvalid, int kvalid) {
  const int col = lane & 15;
  const int kg  = (lane >> 4) & 1;
  v16h b;
#pragma unroll
  for (int h = 0; h < 16; ++h) {
    const int k = ((h >> 3) << 4) + (kg << 3) + (h & 7);
    f16 v = (f16)0.f;
    if (col < cvalid && k < kvalid) v = base[col * cstride + k * kstride];
    b[h] = v;
  }
  return b;
}

// f32 -> f16 converting variants (full 16 rows / 32 K valid)
__device__ __forceinline__ v16h gatherA_f32(const float* base, int rstride, int kstride,
                                            int lane) {
  const int row = lane & 15;
  const int kg  = (lane >> 4) & 1;
  v16h a;
#pragma unroll
  for (int h = 0; h < 16; ++h) {
    const int k = ((h >> 3) << 4) + (kg << 3) + (h & 7);
    a[h] = (f16)base[row * rstride + k * kstride];
  }
  return a;
}

__device__ __forceinline__ v16h gatherB_f32(const float* base, int cstride, int kstride,
                                            int lane) {
  const int col = lane & 15;
  const int kg  = (lane >> 4) & 1;
  v16h b;
#pragma unroll
  for (int h = 0; h < 16; ++h) {
    const int k = ((h >> 3) << 4) + (kg << 3) + (h & 7);
    b[h] = (f16)base[col * cstride + k * kstride];
  }
  return b;
}

// ---------- cross-lane row reductions (rows live within a 16-lane half) ----------
__device__ __forceinline__ float redmax16(float v) {
  v = fmaxf(v, __shfl_xor(v, 1, 32));
  v = fmaxf(v, __shfl_xor(v, 2, 32));
  v = fmaxf(v, __shfl_xor(v, 4, 32));
  v = fmaxf(v, __shfl_xor(v, 8, 32));
  return v;
}
__device__ __forceinline__ float redsum16(float v) {
  v += __shfl_xor(v, 1, 32);
  v += __shfl_xor(v, 2, 32);
  v += __shfl_xor(v, 4, 32);
  v += __shfl_xor(v, 8, 32);
  return v;
}

// ---------- Tensor Data Mover: async 2D f16 tile load Global -> LDS ----------
// Builds D# per CDNA5 ISA (§8.3/§8.4), 2-group form (2D tile).
// tile = tile_d0 (contiguous, elements) x tile_d1 (rows); row stride d0_stride
// elements. LDS receives tile rows densely packed ([row][tile_d0] f16).
__device__ __forceinline__ void tdm_load_2d_f16(uint32_t lds_addr, const f16* gptr,
                                                uint32_t tile_d0, uint32_t tile_d1,
                                                uint32_t tensor_d0, uint32_t tensor_d1,
                                                uint32_t d0_stride) {
  const uint64_t ga = (uint64_t)(uintptr_t)gptr;
  v4i g0;
  g0[0] = 1;                                   // count=1, user descriptor
  g0[1] = (int)lds_addr;                       // lds_addr [63:32]
  g0[2] = (int)(uint32_t)ga;                   // global_addr [95:64]
  g0[3] = (int)(((uint32_t)(ga >> 32) & 0x01FFFFFFu) | (2u << 30)); // addr[56:32]|type=2
  v8i g1;
  g1[0] = (1 << 16);                           // workgroup_mask=0, data_size=1 (2B)
  g1[1] = (int)((tensor_d0 & 0xFFFFu) << 16);  // atomic_barrier=0 | tensor_dim0 lo
  g1[2] = (int)(((tensor_d0 >> 16) & 0xFFFFu) | ((tensor_d1 & 0xFFFFu) << 16));
  g1[3] = (int)(((tensor_d1 >> 16) & 0xFFFFu) | ((tile_d0 & 0xFFFFu) << 16));
  g1[4] = (int)(tile_d1 & 0xFFFFu);            // tile_dim1 | tile_dim2=0
  g1[5] = (int)d0_stride;                      // tensor_dim0_stride[31:0]
  g1[6] = 0;                                   // stride0 hi | dim1_stride lo
  g1[7] = 0;                                   // dim1_stride hi (2D: unused)
  asm volatile("tensor_load_to_lds %0, %1" :: "s"(g0), "s"(g1) : "memory");
}

// ============================================================================
// Kernel 1: fused QKV projection + bias + cos(. + theta) gate, f16 output.
// grid = (BS/16, E/16, 3), block = 32 (one wave -> one 16x16 output tile)
// ============================================================================
__global__ __launch_bounds__(32)
void qkv_cos_kernel(const float* __restrict__ x,
                    const float* __restrict__ Wq, const float* __restrict__ bq,
                    const float* __restrict__ Wk, const float* __restrict__ bk,
                    const float* __restrict__ Wv, const float* __restrict__ bv,
                    const float* __restrict__ theta,
                    f16* __restrict__ qg, f16* __restrict__ kg, f16* __restrict__ vg) {
  const int lane = threadIdx.x;
  const int n0 = blockIdx.x * 16;
  const int j0 = blockIdx.y * 16;
  const float* W;
  const float* bias;
  f16* dst;
  if (blockIdx.z == 0)      { W = Wq; bias = bq; dst = qg; }
  else if (blockIdx.z == 1) { W = Wk; bias = bk; dst = kg; }
  else                      { W = Wv; bias = bv; dst = vg; }

  v8f acc = {};
#pragma unroll
  for (int kb = 0; kb < E_DIM; kb += 32) {
    v16h a = gatherA_f32(x + (size_t)n0 * E_DIM + kb, E_DIM, 1, lane);
    v16h b = gatherB_f32(W + (size_t)j0 * E_DIM + kb, E_DIM, 1, lane);
    acc = wmma_f16(a, b, acc);
  }
  const int n  = lane & 15;
  const int hi = lane >> 4;
  const int j  = j0 + n;
  const float add = bias[j] + theta[j & (DK_DIM - 1)];
#pragma unroll
  for (int r = 0; r < 8; ++r) {
    const int m = n0 + r + (hi << 3);
    dst[(size_t)m * E_DIM + j] = (f16)__cosf(acc[r] + add);
  }
}

// ============================================================================
// Kernel 2: fused flash attention.
// block = 128 (4 waves), each wave owns one 16-query tile of a shared (b,h)
// 64-query supertile. K/V chunks (32 keys x 8 d, f16) are DMA'd into LDS by
// the Tensor Data Mover, double-buffered and overlapped with WMMA compute.
// ============================================================================
#define KCHUNK 32
__global__ __launch_bounds__(128)
void attn_kernel(const f16* __restrict__ qg, const f16* __restrict__ kgm,
                 const f16* __restrict__ vgm, f16* __restrict__ ctx) {
  const int lane = threadIdx.x & 31;
  const int wv   = threadIdx.x >> 5;           // wave id 0..3
  const int blk  = blockIdx.x;                 // B*H*(S/64) = 1024
  const int sbl  = (blk & (S_DIM / 64 - 1)) << 6;
  const int h    = (blk >> 5) & (H_DIM - 1);
  const int b    = blk >> 9;
  const int s0   = sbl + (wv << 4);

  const f16* qbase = qg  + (size_t)(b * S_DIM + s0) * E_DIM + h * DK_DIM;
  const f16* kbase = kgm + (size_t)(b * S_DIM) * E_DIM + h * DK_DIM;
  const f16* vbase = vgm + (size_t)(b * S_DIM) * E_DIM + h * DK_DIM;

  __shared__ f16 kbuf[2][KCHUNK * DK_DIM];     // [t][d], 512 B per buffer
  __shared__ f16 vbuf[2][KCHUNK * DK_DIM];
  __shared__ f16 plds[4][16 * 32];             // per-wave P staging

  const v16h aq = gatherA_f16(qbase, E_DIM, 1, lane, 16, DK_DIM);

  v8f acc = {};
  float mrun[8], lrun[8];
#pragma unroll
  for (int r = 0; r < 8; ++r) { mrun[r] = -3.0e30f; lrun[r] = 0.f; }

  const float scale = 0.35355339059327373f;    // 1/sqrt(DK)
  const int n  = lane & 15;
  const int hi = lane >> 4;

  // Prologue: DMA chunk 0 into buffer 0.
  if (wv == 0) {
    tdm_load_2d_f16((uint32_t)(uintptr_t)&kbuf[0][0], kbase,
                    DK_DIM, KCHUNK, DK_DIM, KCHUNK, E_DIM);
    tdm_load_2d_f16((uint32_t)(uintptr_t)&vbuf[0][0], vbase,
                    DK_DIM, KCHUNK, DK_DIM, KCHUNK, E_DIM);
    __builtin_amdgcn_s_wait_tensorcnt(0);
  }
  __syncthreads();

  for (int t0 = 0; t0 < S_DIM; t0 += KCHUNK) {
    const int cur = (t0 / KCHUNK) & 1;
    // Kick off next chunk's DMA into the other buffer while we compute.
    if (wv == 0 && (t0 + KCHUNK) < S_DIM) {
      tdm_load_2d_f16((uint32_t)(uintptr_t)&kbuf[cur ^ 1][0],
                      kbase + (size_t)(t0 + KCHUNK) * E_DIM,
                      DK_DIM, KCHUNK, DK_DIM, KCHUNK, E_DIM);
      tdm_load_2d_f16((uint32_t)(uintptr_t)&vbuf[cur ^ 1][0],
                      vbase + (size_t)(t0 + KCHUNK) * E_DIM,
                      DK_DIM, KCHUNK, DK_DIM, KCHUNK, E_DIM);
    }

    // scores: B operand from LDS K tile ([t][d], cstride=8, kstride=1)
    const v16h bk0 = gatherB_f16(&kbuf[cur][0],           DK_DIM, 1, lane, 16, DK_DIM);
    const v16h bk1 = gatherB_f16(&kbuf[cur][16 * DK_DIM], DK_DIM, 1, lane, 16, DK_DIM);
    v8f sc0 = {};
    v8f sc1 = {};
    sc0 = wmma_f16(aq, bk0, sc0);
    sc1 = wmma_f16(aq, bk1, sc1);

#pragma unroll
    for (int r = 0; r < 8; ++r) {
      const float a0 = sc0[r] * scale;
      const float a1 = sc1[r] * scale;
      const float mc   = redmax16(fmaxf(a0, a1));
      const float mnew = fmaxf(mrun[r], mc);
      const float corr = __expf(mrun[r] - mnew);
      mrun[r] = mnew;
      const float p0 = __expf(a0 - mnew);
      const float p1 = __expf(a1 - mnew);
      lrun[r] = lrun[r] * corr + redsum16(p0 + p1);
      acc[r] *= corr;
      const int m = r + (hi << 3);
      plds[wv][m * 32 + n]      = (f16)p0;
      plds[wv][m * 32 + 16 + n] = (f16)p1;
    }
    asm volatile("s_wait_dscnt 0" ::: "memory");   // LDS RAW fence (P staging)

    // P (A op) from LDS; V (B op) from LDS V tile: elem(col=d,k=t)=vbuf[t*8+d]
    const v16h pa = gatherA_f16(&plds[wv][0], 32, 1, lane, 16, 32);
    const v16h bv = gatherB_f16(&vbuf[cur][0], 1, DK_DIM, lane, DK_DIM, KCHUNK);
    acc = wmma_f16(pa, bv, acc);

    // Close the pipeline stage: DMA for next buffer must be complete, and all
    // waves must be done reading before it gets overwritten next iteration.
    if (wv == 0) __builtin_amdgcn_s_wait_tensorcnt(0);
    __syncthreads();
  }

  if (n < DK_DIM) {
#pragma unroll
    for (int r = 0; r < 8; ++r) {
      const int m = s0 + r + (hi << 3);
      ctx[(size_t)(b * S_DIM + m) * E_DIM + h * DK_DIM + n] = (f16)(acc[r] / lrun[r]);
    }
  }
}

// ============================================================================
// Kernel 3: output projection ctx(f16) @ Wc^T + bc -> f32
// ============================================================================
__global__ __launch_bounds__(32)
void outproj_kernel(const f16* __restrict__ ctx, const float* __restrict__ Wc,
                    const float* __restrict__ bc, float* __restrict__ out) {
  const int lane = threadIdx.x;
  const int n0 = blockIdx.x * 16;
  const int j0 = blockIdx.y * 16;
  v8f acc = {};
#pragma unroll
  for (int kb = 0; kb < E_DIM; kb += 32) {
    v16h a = gatherA_f16(ctx + (size_t)n0 * E_DIM + kb, E_DIM, 1, lane, 16, 32);
    v16h b = gatherB_f32(Wc + (size_t)j0 * E_DIM + kb, E_DIM, 1, lane);
    acc = wmma_f16(a, b, acc);
  }
  const int n  = lane & 15;
  const int hi = lane >> 4;
  const float bj = bc[j0 + n];
#pragma unroll
  for (int r = 0; r < 8; ++r) {
    const int m = n0 + r + (hi << 3);
    out[(size_t)m * E_DIM + j0 + n] = acc[r] + bj;
  }
}

// ============================================================================
extern "C" void kernel_launch(void* const* d_in, const int* in_sizes, int n_in,
                              void* d_out, int out_size, void* d_ws, size_t ws_size,
                              hipStream_t stream) {
  (void)in_sizes; (void)n_in; (void)out_size; (void)ws_size;
  const float* x  = (const float*)d_in[0];
  const float* Wq = (const float*)d_in[1];
  const float* bq = (const float*)d_in[2];
  const float* Wk = (const float*)d_in[3];
  const float* bk = (const float*)d_in[4];
  const float* Wv = (const float*)d_in[5];
  const float* bv = (const float*)d_in[6];
  const float* th = (const float*)d_in[7];
  const float* Wc = (const float*)d_in[8];
  const float* bc = (const float*)d_in[9];
  float* out = (float*)d_out;

  const size_t buf = (size_t)B_DIM * S_DIM * E_DIM * sizeof(f16);  // 1 MiB each
  char* w = (char*)d_ws;
  f16* qg  = (f16*)(w + 0 * buf);
  f16* kg  = (f16*)(w + 1 * buf);
  f16* vg  = (f16*)(w + 2 * buf);
  f16* ctx = (f16*)(w + 3 * buf);

  dim3 g1((B_DIM * S_DIM) / 16, E_DIM / 16, 3);
  qkv_cos_kernel<<<g1, 32, 0, stream>>>(x, Wq, bq, Wk, bk, Wv, bv, th, qg, kg, vg);

  dim3 g2(B_DIM * H_DIM * (S_DIM / 64));
  attn_kernel<<<g2, 128, 0, stream>>>(qg, kg, vg, ctx);

  dim3 g3((B_DIM * S_DIM) / 16, E_DIM / 16);
  outproj_kernel<<<g3, 32, 0, stream>>>(ctx, Wc, bc, out);
}
